// Mamba3LMHeadModel_9234179686666
// MI455X (gfx1250) — compile-verified, matrix operand order checked by reading, not run on hardware
//
#include <hip/hip_runtime.h>
#include <math.h>

// ---------------- problem constants ----------------
#define B_SZ    2
#define SEQ     4096
#define DMODEL  2048
#define DINNER  4096
#define NHEADS  64
#define HEADDIM 64
#define DSTATE  128
#define CHUNK   64
#define NCHUNK  (SEQ / CHUNK)       // 64
#define DPROJ   8640                // 2*4096 + 2*128 + 2*64 + 64

// zxbcdt column offsets
#define OFF_Z   0
#define OFF_X   4096
#define OFF_B   8192
#define OFF_C   8320
#define OFF_DT  8448
#define OFF_LAM 8512
#define OFF_TH  8576

typedef __attribute__((ext_vector_type(16))) __bf16 v16bf;
typedef __attribute__((ext_vector_type(8)))  __bf16 v8bf;
typedef __attribute__((ext_vector_type(8)))  float  v8f;
typedef __attribute__((ext_vector_type(4)))  unsigned int u32x4;
typedef __attribute__((ext_vector_type(8)))  int i32x8;
typedef __attribute__((ext_vector_type(4)))  int i32x4;

// fill 8 consecutive v16bf elements from 8 bf16 (16B aligned)
__device__ __forceinline__ void cp8(const __bf16* __restrict__ p, v16bf& v, int e0) {
    v8bf t = *(const v8bf*)p;
#pragma unroll
    for (int i = 0; i < 8; ++i) v[e0 + i] = t[i];
}

// =====================================================================
// TDM: async 2D tile load global->LDS (bf16 elements).
// D# layout per CDNA5 ISA.
// Pads 4 dwords after every 16 dwords stored -> LDS row stride = tile_d0
// elements + 8 (for tile_d0 == 32: 40 elements = 80B, 16B aligned).
// =====================================================================
#define TDM_ROWSTRIDE 40   // elements per LDS row for 32-wide bf16 tiles

__device__ __forceinline__ void tdm_load_2d(unsigned lds_off, const void* gptr,
                                            unsigned tensor_d0, unsigned tensor_d1,
                                            unsigned tile_d0, unsigned tile_d1,
                                            unsigned stride0_elems) {
    unsigned long long ga = (unsigned long long)gptr;
    u32x4 g0;
    g0[0] = 1u;                                       // count=1 valid descriptor
    g0[1] = lds_off;                                  // lds_addr (bytes)
    g0[2] = (unsigned)(ga & 0xffffffffu);             // global_addr[31:0]
    g0[3] = (unsigned)((ga >> 32) & 0x01ffffffu) |    // global_addr[56:32]
            (2u << 30);                               // type=2 ("image")
    i32x8 g1;
    g1[0] = (int)((1u << 16) |                        // data_size=1 -> 2 bytes
                  (1u << 20) |                        // pad_enable
                  (3u << 22) |                        // pad_interval: 16 dwords
                  (3u << 25));                        // pad_amount: 4 dwords
    g1[1] = (int)((tensor_d0 & 0xffffu) << 16);                        // td0[15:0]
    g1[2] = (int)(((tensor_d0 >> 16) & 0xffffu) |
                  ((tensor_d1 & 0xffffu) << 16));                      // td0[31:16], td1[15:0]
    g1[3] = (int)(((tensor_d1 >> 16) & 0xffffu) |
                  ((tile_d0 & 0xffffu) << 16));                        // td1[31:16], tile_dim0
    g1[4] = (int)(tile_d1 & 0xffffu);                                  // tile_dim1 (tile_dim2=0)
    g1[5] = (int)stride0_elems;                                        // dim0 stride [31:0]
    g1[6] = 0;                                                         // stride bits + dim1 stride
    g1[7] = 0;
    i32x4 z4 = {0, 0, 0, 0};
    i32x8 z8 = {0, 0, 0, 0, 0, 0, 0, 0};
    __builtin_amdgcn_tensor_load_to_lds(g0, g1, z4, z4, z8, 0);
}

// =====================================================================
// fp32 -> bf16 bulk convert (n multiple of 8)
// =====================================================================
__global__ void cvt_bf16_kernel(const float* __restrict__ src, __bf16* __restrict__ dst,
                                long n) {
    long i = ((long)blockIdx.x * blockDim.x + threadIdx.x) * 8;
    if (i + 8 > n) return;
    float4 a = *(const float4*)(src + i);
    float4 b = *(const float4*)(src + i + 4);
    v8bf o;
    o[0] = (__bf16)a.x; o[1] = (__bf16)a.y; o[2] = (__bf16)a.z; o[3] = (__bf16)a.w;
    o[4] = (__bf16)b.x; o[5] = (__bf16)b.y; o[6] = (__bf16)b.z; o[7] = (__bf16)b.w;
    *(v8bf*)(dst + i) = o;
}

// =====================================================================
// TDM-staged GEMM: C[M,N] = A[M,K] * W[N,K]^T, bf16 WMMA, f32 acc.
// Block = 256 threads (8 waves) -> 128 rows x 64 cols per block.
// Per 32-wide k-step: TDM loads A tile (128x32) + W tile (64x32) into
// double-buffered LDS; wave w computes row-tile w against all 4 col-tiles.
// =====================================================================
__device__ __forceinline__ void gemm_tile_step(const __bf16* Asl, const __bf16* Wsl,
                                               int wave, int lidx, int koff, v8f acc[4]) {
    const __bf16* Ar = Asl + (size_t)(wave * 16 + lidx) * TDM_ROWSTRIDE;
    v16bf a;
    cp8(Ar + koff,      a, 0);
    cp8(Ar + 16 + koff, a, 8);
#pragma unroll
    for (int j = 0; j < 4; ++j) {
        const __bf16* Wr = Wsl + (size_t)(j * 16 + lidx) * TDM_ROWSTRIDE;
        v16bf bb;
        cp8(Wr + koff,      bb, 0);
        cp8(Wr + 16 + koff, bb, 8);
        acc[j] = __builtin_amdgcn_wmma_f32_16x16x32_bf16(false, a, false, bb,
                                                         (short)0, acc[j], false, false);
    }
}

__global__ void gemm_wmma_tdm(const __bf16* __restrict__ A, const __bf16* __restrict__ W,
                              float* __restrict__ Cout, int M, int N, int K) {
    __shared__ __bf16 Asl[2][128 * TDM_ROWSTRIDE];
    __shared__ __bf16 Wsl[2][64 * TDM_ROWSTRIDE];

    int lane = threadIdx.x & 31;
    int wave = threadIdx.x >> 5;
    int lidx = lane & 15;
    int koff = (lane >> 4) * 8;

    int row0 = blockIdx.x * 128;
    int col0 = blockIdx.y * 64;

    v8f acc[4] = {{}, {}, {}, {}};

    // prologue: fill buffer 0
    if (wave == 0) {
        tdm_load_2d((unsigned)(size_t)&Asl[0][0], A + (size_t)row0 * K,
                    K, M - row0, 32, 128, K);
        tdm_load_2d((unsigned)(size_t)&Wsl[0][0], W + (size_t)col0 * K,
                    K, N - col0, 32, 64, K);
    }

    int cur = 0;
    for (int k0 = 0; k0 < K - 32; k0 += 32) {
        if (wave == 0) {
            // issue next buffer, then wait for the two oldest (current buffer)
            tdm_load_2d((unsigned)(size_t)&Asl[cur ^ 1][0],
                        A + (size_t)row0 * K + k0 + 32,
                        K - (k0 + 32), M - row0, 32, 128, K);
            tdm_load_2d((unsigned)(size_t)&Wsl[cur ^ 1][0],
                        W + (size_t)col0 * K + k0 + 32,
                        K - (k0 + 32), N - col0, 32, 64, K);
            __builtin_amdgcn_s_wait_tensorcnt(2);
        }
        __syncthreads();
        gemm_tile_step(&Asl[cur][0], &Wsl[cur][0], wave, lidx, koff, acc);
        __syncthreads();   // everyone done reading cur before it is re-filled
        cur ^= 1;
    }
    if (wave == 0) __builtin_amdgcn_s_wait_tensorcnt(0);
    __syncthreads();
    gemm_tile_step(&Asl[cur][0], &Wsl[cur][0], wave, lidx, koff, acc);

    int rbase = (lane >> 4) * 8;
#pragma unroll
    for (int j = 0; j < 4; ++j) {
        int cn = col0 + j * 16 + lidx;
#pragma unroll
        for (int r = 0; r < 8; ++r) {
            Cout[(size_t)(row0 + wave * 16 + rbase + r) * N + cn] = acc[j][r];
        }
    }
}

// =====================================================================
// prep: dt = softplus(dtr + bias); lam = sigmoid; Aval = -exp(A_log)*dt;
//       xbf = bf16(x segment).  grid = B*SEQ, block = 256
// =====================================================================
__global__ void prep_kernel(const float* __restrict__ zx, const float* __restrict__ A_log,
                            const float* __restrict__ dt_bias,
                            float* __restrict__ dt, float* __restrict__ lam,
                            float* __restrict__ Aval, __bf16* __restrict__ xbf) {
    int row = blockIdx.x;  // b*SEQ + l
    const float* zr = zx + (size_t)row * DPROJ;
    int tid = threadIdx.x;
    if (tid < 64) {
        float d  = zr[OFF_DT + tid] + dt_bias[tid];
        float sp = (d > 20.f) ? d : log1pf(__expf(d));
        dt[(size_t)row * 64 + tid]  = sp;
        float lr = zr[OFF_LAM + tid];
        lam[(size_t)row * 64 + tid] = 1.f / (1.f + __expf(-lr));
        Aval[(size_t)row * 64 + tid] = -__expf(A_log[tid]) * sp;
    }
#pragma unroll
    for (int i = 0; i < 16; ++i) {
        int c = tid * 16 + i;
        xbf[(size_t)row * DINNER + c] = (__bf16)zr[OFF_X + c];
    }
}

// =====================================================================
// ang = cumsum_l(theta)  per (b, ch).  grid = B*64, block = 256
// =====================================================================
__global__ void angscan_kernel(const float* __restrict__ zx, float* __restrict__ ang) {
    int b  = blockIdx.x >> 6;
    int ch = blockIdx.x & 63;
    int tid = threadIdx.x;
    __shared__ float part[256];
    float loc[16];
    float s = 0.f;
    int l0 = tid * 16;
#pragma unroll
    for (int i = 0; i < 16; ++i) {
        int l   = l0 + i;
        float v = zx[((size_t)(b * SEQ + l)) * DPROJ + OFF_TH + ch];
        s += v;
        loc[i] = s;
    }
    part[tid] = s;
    __syncthreads();
    for (int off = 1; off < 256; off <<= 1) {
        float v   = part[tid];
        float add = (tid >= off) ? part[tid - off] : 0.f;
        __syncthreads();
        part[tid] = v + add;
        __syncthreads();
    }
    float base = (tid > 0) ? part[tid - 1] : 0.f;
#pragma unroll
    for (int i = 0; i < 16; ++i) {
        int l = l0 + i;
        ang[((size_t)(b * SEQ + l)) * 64 + ch] = base + loc[i];
    }
}

// =====================================================================
// rope: Bn = rope(rms(Braw)), Cn = rope(rms(Craw)), Bsh = shift(Bn)
// grid = B*SEQ, block = 128
// =====================================================================
__global__ void rope_kernel(const float* __restrict__ zx, const float* __restrict__ ang,
                            const float* __restrict__ Bw, const float* __restrict__ Cw,
                            __bf16* __restrict__ Bn, __bf16* __restrict__ Cn,
                            __bf16* __restrict__ Bsh) {
    int row = blockIdx.x;  // b*SEQ + l
    int l   = row % SEQ;
    int t   = threadIdx.x;  // 0..127
    __shared__ float sq[128];
    __shared__ float bv[128], cv[128];
    const float* zr = zx + (size_t)row * DPROJ;
    float xb = zr[OFF_B + t], xc = zr[OFF_C + t];

    sq[t] = xb * xb; __syncthreads();
    for (int off = 64; off >= 1; off >>= 1) { if (t < off) sq[t] += sq[t + off]; __syncthreads(); }
    float rb = rsqrtf(sq[0] / 128.f + 1e-5f);
    __syncthreads();
    sq[t] = xc * xc; __syncthreads();
    for (int off = 64; off >= 1; off >>= 1) { if (t < off) sq[t] += sq[t + off]; __syncthreads(); }
    float rc = rsqrtf(sq[0] / 128.f + 1e-5f);
    __syncthreads();

    bv[t] = xb * rb * Bw[t];
    cv[t] = xc * rc * Cw[t];
    __syncthreads();

    int   i  = t >> 1;
    float a  = ang[(size_t)row * 64 + i];
    float ca = __cosf(a), sa = __sinf(a);
    float ob, oc;
    if ((t & 1) == 0) { ob = ca * bv[t] - sa * bv[t + 1]; oc = ca * cv[t] - sa * cv[t + 1]; }
    else              { ob = sa * bv[t - 1] + ca * bv[t]; oc = sa * cv[t - 1] + ca * cv[t]; }
    Bn[(size_t)row * DSTATE + t] = (__bf16)ob;
    Cn[(size_t)row * DSTATE + t] = (__bf16)oc;
    if (l + 1 < SEQ) Bsh[((size_t)row + 1) * DSTATE + t] = (__bf16)ob;
    if (l == 0)      Bsh[(size_t)row * DSTATE + t]       = (__bf16)0.f;
}

// =====================================================================
// A_cs: per-chunk inclusive cumsum.  grid = B*NHEADS*NCHUNK, block = 64
// =====================================================================
__global__ void acs_kernel(const float* __restrict__ Aval, float* __restrict__ Acs,
                           float* __restrict__ Tsum) {
    int idx = blockIdx.x;
    int c = idx & 63, h = (idx >> 6) & 63, b = idx >> 12;
    int t = threadIdx.x;
    __shared__ float s[64];
    int l = c * CHUNK + t;
    s[t]  = Aval[((size_t)(b * SEQ + l)) * 64 + h];
    __syncthreads();
    for (int off = 1; off < 64; off <<= 1) {
        float v = s[t];
        float add = (t >= off) ? s[t - off] : 0.f;
        __syncthreads();
        s[t] = v + add;
        __syncthreads();
    }
    Acs[((size_t)(b * NHEADS + h)) * SEQ + l] = s[t];
    if (t == 63) Tsum[((size_t)(b * NHEADS + h)) * NCHUNK + c] = s[63];
}

// =====================================================================
// SSD diagonal kernel: per (b, chunk, head), both variants.
// Tile map per wave: mt = wave>>1, nt = (wave&1)*2 + tt  (A operand shared)
// grid = B*NCHUNK*NHEADS, block = 256 (8 waves)
// =====================================================================
__global__ void ssd_diag_kernel(const __bf16* __restrict__ Bn, const __bf16* __restrict__ Bsh,
                                const __bf16* __restrict__ Cn, const __bf16* __restrict__ xbf,
                                const float* __restrict__ dt, const float* __restrict__ lam,
                                const float* __restrict__ Acs, __bf16* __restrict__ states,
                                float* __restrict__ ybuf) {
    int idx = blockIdx.x;
    int h = idx & 63, c = (idx >> 6) & 63, b = idx >> 12;
    int tid = threadIdx.x, lane = tid & 31, wave = tid >> 5;
    int lbase = c * CHUNK;
    size_t rowbase = (size_t)b * SEQ + lbase;

    __shared__ __bf16 Gs[64][72];    // [l][s] masked C.B^T
    __shared__ __bf16 XsT[64][72];   // [p][s] scaled x
    __shared__ __bf16 BdT[128][72];  // [n][s] B * decay
    __shared__ float AcsS[64], decayS[64];

    const float* AcsRow = Acs + ((size_t)(b * NHEADS + h)) * SEQ + lbase;
    if (tid < 64) AcsS[tid] = AcsRow[tid];
    __syncthreads();
    if (tid < 64) decayS[tid] = __expf(AcsS[63] - AcsS[tid]);
    __syncthreads();

    int koff = (lane >> 4) * 8;
    int lidx = lane & 15;
    int rb   = (lane >> 4) * 8;
    int mt   = wave >> 1;           // shared M-tile for this wave's 2 tiles
    int ntb  = (wave & 1) * 2;      // base N-tile

    v8f yacc[2] = {{}, {}};  // persists across both variants

    for (int v = 0; v < 2; ++v) {
        const __bf16* Bv = v ? Bsh : Bn;

        // ---- phase 1: G = C Bv^T  (A hoisted across both tiles) ----
        {
            const __bf16* Ca = Cn + (rowbase + mt * 16 + lidx) * DSTATE;
            v8f gacc[2] = {{}, {}};
#pragma unroll
            for (int k0 = 0; k0 < 128; k0 += 32) {
                v16bf a;
                cp8(Ca + k0 + koff,      a, 0);
                cp8(Ca + k0 + 16 + koff, a, 8);
#pragma unroll
                for (int tt = 0; tt < 2; ++tt) {
                    const __bf16* Ba = Bv + (rowbase + (ntb + tt) * 16 + lidx) * DSTATE;
                    v16bf bb;
                    cp8(Ba + k0 + koff,      bb, 0);
                    cp8(Ba + k0 + 16 + koff, bb, 8);
                    gacc[tt] = __builtin_amdgcn_wmma_f32_16x16x32_bf16(
                        false, a, false, bb, (short)0, gacc[tt], false, false);
                }
            }
#pragma unroll
            for (int tt = 0; tt < 2; ++tt) {
#pragma unroll
                for (int r = 0; r < 8; ++r) {
                    int ll = mt * 16 + rb + r, ss = (ntb + tt) * 16 + lidx;
                    float g = gacc[tt][r];
                    g = (ss <= ll) ? g * __expf(AcsS[ll] - AcsS[ss]) : 0.f;
                    Gs[ll][ss] = (__bf16)g;
                }
            }
        }

        // ---- stage XsT[p][s] and BdT[n][s] (contiguous global reads) ----
        {
            int s  = tid >> 2;               // 0..63
            int lg = lbase + s - v;          // v=1: x[l-1]
            // XsT: 16 p's per thread
            int p0 = (tid & 3) * 16;
            if (lg >= 0) {
                float dtv = dt[((size_t)b * SEQ + lg) * 64 + h];
                float lm  = lam[((size_t)b * SEQ + lg) * 64 + h];
                float sc  = dtv * (v ? (1.f - lm) : lm);
                const __bf16* xr = xbf + ((size_t)b * SEQ + lg) * DINNER + h * HEADDIM + p0;
                v8bf t0 = *(const v8bf*)(xr);
                v8bf t1 = *(const v8bf*)(xr + 8);
#pragma unroll
                for (int i = 0; i < 8; ++i) XsT[p0 + i][s]     = (__bf16)((float)t0[i] * sc);
#pragma unroll
                for (int i = 0; i < 8; ++i) XsT[p0 + 8 + i][s] = (__bf16)((float)t1[i] * sc);
            } else {
#pragma unroll
                for (int i = 0; i < 16; ++i) XsT[p0 + i][s] = (__bf16)0.f;
            }
            // BdT: 32 n's per thread
            int n0 = (tid & 3) * 32;
            float dsc = decayS[s];
            const __bf16* br = Bv + (rowbase + s) * DSTATE + n0;
#pragma unroll
            for (int q = 0; q < 4; ++q) {
                v8bf t = *(const v8bf*)(br + q * 8);
#pragma unroll
                for (int i = 0; i < 8; ++i)
                    BdT[n0 + q * 8 + i][s] = (__bf16)((float)t[i] * dsc);
            }
        }
        __syncthreads();

        // ---- phase 2: Y += G @ X (A hoisted) ----
#pragma unroll
        for (int k0 = 0; k0 < 64; k0 += 32) {
            v16bf a;
            cp8(&Gs[mt * 16 + lidx][k0 + koff],      a, 0);
            cp8(&Gs[mt * 16 + lidx][k0 + 16 + koff], a, 8);
#pragma unroll
            for (int tt = 0; tt < 2; ++tt) {
                int pcol = (ntb + tt) * 16 + lidx;
                v16bf bb;
                cp8(&XsT[pcol][k0 + koff],      bb, 0);
                cp8(&XsT[pcol][k0 + 16 + koff], bb, 8);
                yacc[tt] = __builtin_amdgcn_wmma_f32_16x16x32_bf16(
                    false, a, false, bb, (short)0, yacc[tt], false, false);
            }
        }

        // ---- phase 3: chunk states S[p][n] = Bd^T @ X (ntile = wave) ----
        {
            int nrow = wave * 16 + lidx;
            v8f sacc[4] = {{}, {}, {}, {}};
#pragma unroll
            for (int k0 = 0; k0 < 64; k0 += 32) {
                v16bf a;
                cp8(&BdT[nrow][k0 + koff],      a, 0);
                cp8(&BdT[nrow][k0 + 16 + koff], a, 8);
#pragma unroll
                for (int tt = 0; tt < 4; ++tt) {
                    int pcol = tt * 16 + lidx;
                    v16bf bb;
                    cp8(&XsT[pcol][k0 + koff],      bb, 0);
                    cp8(&XsT[pcol][k0 + 16 + koff], bb, 8);
                    sacc[tt] = __builtin_amdgcn_wmma_f32_16x16x32_bf16(
                        false, a, false, bb, (short)0, sacc[tt], false, false);
                }
            }
            size_t sb = (((size_t)v * B_SZ + b) * NHEADS + h) * NCHUNK + c;
#pragma unroll
            for (int tt = 0; tt < 4; ++tt) {
                int pcol = tt * 16 + lidx;
#pragma unroll
                for (int r = 0; r < 8; ++r) {
                    int n = wave * 16 + rb + r;
                    states[(sb * HEADDIM + pcol) * DSTATE + n] = (__bf16)sacc[tt][r];
                }
            }
        }
        __syncthreads();  // LDS reused by next variant
    }

    // ---- store Y_diag (y1+y2) ----
#pragma unroll
    for (int tt = 0; tt < 2; ++tt) {
        int pcol = (ntb + tt) * 16 + lidx;
#pragma unroll
        for (int r = 0; r < 8; ++r) {
            int l = lbase + mt * 16 + rb + r;
            ybuf[(((size_t)b * SEQ + l) * NHEADS + h) * HEADDIM + pcol] = yacc[tt][r];
        }
    }
}

// =====================================================================
// inter-chunk recurrence (in place): read S[c], write inter[c]=carry,
// carry = carry*exp(Tsum[c]) + S[c].  grid = 2*B*NHEADS, block = 256
// =====================================================================
__global__ void chunkscan_kernel(__bf16* __restrict__ states, const float* __restrict__ Tsum) {
    int idx = blockIdx.x;
    int h = idx & 63, b = (idx >> 6) & 1, v = idx >> 7;
    int tid = threadIdx.x;
    size_t base = ((size_t)v * B_SZ + b) * NHEADS + h;
    base *= NCHUNK;
    const float* Tr = Tsum + ((size_t)b * NHEADS + h) * NCHUNK;
    float carry[32];
#pragma unroll
    for (int i = 0; i < 32; ++i) carry[i] = 0.f;
    for (int c = 0; c < NCHUNK; ++c) {
        __bf16* ptr = states + (base + c) * (HEADDIM * DSTATE) + (size_t)tid * 32;
        float sv[32];
#pragma unroll
        for (int i = 0; i < 32; ++i) sv[i] = (float)ptr[i];
#pragma unroll
        for (int i = 0; i < 32; ++i) ptr[i] = (__bf16)carry[i];
        float dk = __expf(Tr[c]);
#pragma unroll
        for (int i = 0; i < 32; ++i) carry[i] = carry[i] * dk + sv[i];
    }
}

// =====================================================================
// Y_off: y += (C * exp(Acs)) @ inter, both variants.
// grid = B*NCHUNK*NHEADS, block = 256
// =====================================================================
__global__ void yoff_kernel(const __bf16* __restrict__ Cn, const __bf16* __restrict__ states,
                            const float* __restrict__ Acs, float* __restrict__ ybuf) {
    int idx = blockIdx.x;
    int h = idx & 63, c = (idx >> 6) & 63, b = idx >> 12;
    int tid = threadIdx.x, lane = tid & 31, wave = tid >> 5;
    int lidx = lane & 15, koff = (lane >> 4) * 8;
    int lbase = c * CHUNK;
    int mt  = wave >> 1;
    int ntb = (wave & 1) * 2;
    const float* AcsRow = Acs + ((size_t)(b * NHEADS + h)) * SEQ + lbase;

    int lrow   = mt * 16 + lidx;
    float ecs  = __expf(AcsRow[lrow]);
    const __bf16* Ca = Cn + ((size_t)b * SEQ + lbase + lrow) * DSTATE;

    v8f acc[2] = {{}, {}};
#pragma unroll
    for (int k0 = 0; k0 < 128; k0 += 32) {
        v16bf a;
        {
            v8bf c0 = *(const v8bf*)(Ca + k0 + koff);
            v8bf c1 = *(const v8bf*)(Ca + k0 + 16 + koff);
#pragma unroll
            for (int e = 0; e < 8; ++e) a[e]     = (__bf16)((float)c0[e] * ecs);
#pragma unroll
            for (int e = 0; e < 8; ++e) a[8 + e] = (__bf16)((float)c1[e] * ecs);
        }
#pragma unroll
        for (int v = 0; v < 2; ++v) {
            size_t sb = (((size_t)v * B_SZ + b) * NHEADS + h) * NCHUNK + c;
#pragma unroll
            for (int tt = 0; tt < 2; ++tt) {
                int pcol = (ntb + tt) * 16 + lidx;
                const __bf16* Ip = states + sb * (HEADDIM * DSTATE) + (size_t)pcol * DSTATE;
                v16bf bb;
                cp8(Ip + k0 + koff,      bb, 0);
                cp8(Ip + k0 + 16 + koff, bb, 8);
                acc[tt] = __builtin_amdgcn_wmma_f32_16x16x32_bf16(
                    false, a, false, bb, (short)0, acc[tt], false, false);
            }
        }
    }
    int rb = (lane >> 4) * 8;
#pragma unroll
    for (int tt = 0; tt < 2; ++tt) {
        int pcol = (ntb + tt) * 16 + lidx;
#pragma unroll
        for (int r = 0; r < 8; ++r) {
            int l    = lbase + mt * 16 + rb + r;
            size_t o = (((size_t)b * SEQ + l) * NHEADS + h) * HEADDIM + pcol;
            ybuf[o] += acc[tt][r];
        }
    }
}

// =====================================================================
// gating + RMS norm -> yn (bf16).  grid = B*SEQ, block = 256
// =====================================================================
__global__ void gate_kernel(const float* __restrict__ zx, const float* __restrict__ ybuf,
                            const __bf16* __restrict__ xbf, const float* __restrict__ Dp,
                            const float* __restrict__ wnorm, __bf16* __restrict__ yn) {
    int row = blockIdx.x;
    int tid = threadIdx.x;
    __shared__ float red[256];
    const float* zr = zx + (size_t)row * DPROJ;
    float tv[16];
    float ss = 0.f;
#pragma unroll
    for (int i = 0; i < 16; ++i) {
        int cc = tid * 16 + i;
        int h  = cc >> 6;
        float y = ybuf[(size_t)row * DINNER + cc] +
                  Dp[h] * (float)xbf[(size_t)row * DINNER + cc];
        float z = zr[OFF_Z + cc];
        float g = z / (1.f + __expf(-z));
        float t = y * g;
        tv[i] = t;
        ss += t * t;
    }
    red[tid] = ss;
    __syncthreads();
    for (int off = 128; off >= 1; off >>= 1) {
        if (tid < off) red[tid] += red[tid + off];
        __syncthreads();
    }
    float scale = rsqrtf(red[0] / (float)DINNER + 1e-5f);
#pragma unroll
    for (int i = 0; i < 16; ++i) {
        int cc = tid * 16 + i;
        yn[(size_t)row * DINNER + cc] = (__bf16)(tv[i] * scale * wnorm[cc]);
    }
}

// =====================================================================
// host launcher
// =====================================================================
extern "C" void kernel_launch(void* const* d_in, const int* in_sizes, int n_in,
                              void* d_out, int out_size, void* d_ws, size_t ws_size,
                              hipStream_t stream) {
    const float* u        = (const float*)d_in[0];
    const float* W_in     = (const float*)d_in[1];
    const float* A_log    = (const float*)d_in[2];
    const float* Dp       = (const float*)d_in[3];
    const float* dt_bias  = (const float*)d_in[4];
    const float* B_norm_w = (const float*)d_in[5];
    const float* C_norm_w = (const float*)d_in[6];
    const float* out_norm = (const float*)d_in[7];
    const float* W_out    = (const float*)d_in[8];
    float* out = (float*)d_out;

    // workspace carve-up (256B aligned)
    char*  ws  = (char*)d_ws;
    size_t off = 0;
    auto alloc = [&](size_t bytes) {
        size_t o = off;
        off += (bytes + 255) & ~(size_t)255;
        return (void*)(ws + o);
    };
    float*  zx     = (float*) alloc((size_t)B_SZ * SEQ * DPROJ * 4);
    __bf16* xbf    = (__bf16*)alloc((size_t)B_SZ * SEQ * DINNER * 2);
    float*  dt     = (float*) alloc((size_t)B_SZ * SEQ * 64 * 4);
    float*  lam    = (float*) alloc((size_t)B_SZ * SEQ * 64 * 4);
    float*  Aval   = (float*) alloc((size_t)B_SZ * SEQ * 64 * 4);
    float*  ang    = (float*) alloc((size_t)B_SZ * SEQ * 64 * 4);
    __bf16* Bn     = (__bf16*)alloc((size_t)B_SZ * SEQ * DSTATE * 2);
    __bf16* Cn     = (__bf16*)alloc((size_t)B_SZ * SEQ * DSTATE * 2);
    __bf16* Bsh    = (__bf16*)alloc((size_t)B_SZ * SEQ * DSTATE * 2);
    float*  Acs    = (float*) alloc((size_t)B_SZ * NHEADS * SEQ * 4);
    float*  Tsum   = (float*) alloc((size_t)B_SZ * NHEADS * NCHUNK * 4);
    __bf16* states = (__bf16*)alloc((size_t)2 * B_SZ * NHEADS * NCHUNK * HEADDIM * DSTATE * 2);
    float*  ybuf   = (float*) alloc((size_t)B_SZ * SEQ * NHEADS * HEADDIM * 4);
    __bf16* yn     = (__bf16*)alloc((size_t)B_SZ * SEQ * DINNER * 2);
    __bf16* ubf    = (__bf16*)alloc((size_t)B_SZ * SEQ * DMODEL * 2);
    __bf16* Wibf   = (__bf16*)alloc((size_t)DPROJ * DMODEL * 2);
    __bf16* Wobf   = (__bf16*)alloc((size_t)DMODEL * DINNER * 2);
    (void)ws_size; (void)in_sizes; (void)n_in; (void)out_size;

    const int M = B_SZ * SEQ;  // 8192

    // 0) one-shot bf16 conversions of GEMM operands
    {
        long nu = (long)B_SZ * SEQ * DMODEL;
        long ni = (long)DPROJ * DMODEL;
        long no = (long)DMODEL * DINNER;
        cvt_bf16_kernel<<<(int)(nu / 8 / 256), 256, 0, stream>>>(u, ubf, nu);
        cvt_bf16_kernel<<<(int)((ni / 8 + 255) / 256), 256, 0, stream>>>(W_in, Wibf, ni);
        cvt_bf16_kernel<<<(int)(no / 8 / 256), 256, 0, stream>>>(W_out, Wobf, no);
    }

    // 1) in-projection GEMM (TDM-staged): zx = ubf @ Wibf^T  (8192 x 8640 x 2048)
    gemm_wmma_tdm<<<dim3(M / 128, DPROJ / 64), 256, 0, stream>>>(
        ubf, Wibf, zx, M, DPROJ, DMODEL);

    // 2) elementwise prep
    prep_kernel<<<M, 256, 0, stream>>>(zx, A_log, dt_bias, dt, lam, Aval, xbf);

    // 3) theta cumsum over sequence
    angscan_kernel<<<B_SZ * 64, 256, 0, stream>>>(zx, ang);

    // 4) RMS + RoPE for B and C, plus shifted B
    rope_kernel<<<M, 128, 0, stream>>>(zx, ang, B_norm_w, C_norm_w, Bn, Cn, Bsh);

    // 5) per-chunk A cumsum
    acs_kernel<<<B_SZ * NHEADS * NCHUNK, 64, 0, stream>>>(Aval, Acs, Tsum);

    // 6) diagonal SSD (both variants), writes ybuf and chunk states
    ssd_diag_kernel<<<B_SZ * NCHUNK * NHEADS, 256, 0, stream>>>(
        Bn, Bsh, Cn, xbf, dt, lam, Acs, states, ybuf);

    // 7) inter-chunk state recurrence (in place: states -> inter)
    chunkscan_kernel<<<2 * B_SZ * NHEADS, 256, 0, stream>>>(states, Tsum);

    // 8) off-diagonal contribution
    yoff_kernel<<<B_SZ * NCHUNK * NHEADS, 256, 0, stream>>>(Cn, states, Acs, ybuf);

    // 9) gating + RMS norm -> bf16
    gate_kernel<<<M, 256, 0, stream>>>(zx, ybuf, xbf, Dp, out_norm, yn);

    // 10) out-projection GEMM (TDM-staged): out = yn @ Wobf^T (8192 x 2048 x 4096)
    gemm_wmma_tdm<<<dim3(M / 128, DMODEL / 64), 256, 0, stream>>>(
        yn, Wobf, out, M, DMODEL, DINNER);
}
